// MinGRULayer_11957188952158
// MI455X (gfx1250) — matrix operand order repaired and checked by reading, not compiled, not run on hardware
//
#include <hip/hip_runtime.h>

#define B_ 8
#define T_ 4096
#define I_ 1024
#define H_ 1024
#define O_ 1024

typedef __attribute__((ext_vector_type(16))) __bf16 v16bf;
typedef __attribute__((ext_vector_type(8)))  float  v8f;

union Frag {
    v16bf v;
    uint4 q[2];
};

__device__ __forceinline__ unsigned short f32_bf16_rne(float f) {
    unsigned int u = __float_as_uint(f);
    u += 0x7FFFu + ((u >> 16) & 1u);   // round-to-nearest-even
    return (unsigned short)(u >> 16);
}

// ---------------------------------------------------------------------------
// Phase 0: fp32 -> bf16 conversion
// ---------------------------------------------------------------------------
__global__ void cvt_f32_to_bf16(const float* __restrict__ in,
                                unsigned short* __restrict__ out, int n) {
    int i = blockIdx.x * blockDim.x + threadIdx.x;
    if (i < n) out[i] = f32_bf16_rne(in[i]);
}

// ---------------------------------------------------------------------------
// Phase 1: fused (xs@Wz^T + bz -> sigmoid, xs@Wh^T + bh) GEMM + linear scan.
// One wave owns a 16-wide slice of H for one batch element and marches T in
// 16-row tiles, carrying h_{t-1} in registers across tiles.
// Block = 128 threads = 4 waves, each wave a distinct 16-column H slice.
// ---------------------------------------------------------------------------
__global__ __launch_bounds__(128)
void fused_zu_scan(const unsigned short* __restrict__ xsb,   // [B,T,I] bf16
                   const unsigned short* __restrict__ Wzb,   // [H,I]  bf16
                   const unsigned short* __restrict__ Whb,   // [H,I]  bf16
                   const float* __restrict__ bz,
                   const float* __restrict__ bh,
                   unsigned short* __restrict__ hsb) {       // [B,T,H] bf16
    const int lane = threadIdx.x & 31;
    const int wave = threadIdx.x >> 5;
    const int b    = blockIdx.y;
    const int h0   = blockIdx.x * 64 + wave * 16;
    const int n    = lane & 15;   // column within tile (also A row M)
    const int g    = lane >> 4;   // lane half

    const float bzv = bz[h0 + n];
    const float bhv = bh[h0 + n];

    const unsigned short* xbase = xsb + (size_t)b * T_ * I_;
    const unsigned short* wzrow = Wzb + (size_t)(h0 + n) * I_;
    const unsigned short* whrow = Whb + (size_t)(h0 + n) * I_;

    float carry = 0.0f;   // h_{t-1} for column h0+n (valid in all lanes)

    for (int t0 = 0; t0 < T_; t0 += 16) {
        v8f accZ = {}, accU = {};
        const unsigned short* arow = xbase + (size_t)(t0 + n) * I_;

        for (int k = 0; k < I_; k += 32) {
            Frag a, wz, wh;
            // A fragment: 16x32 bf16, lane holds row M=n, K = g*16 + j (contiguous)
            a.q[0] = *(const uint4*)(arow + k + g * 16);
            a.q[1] = *(const uint4*)(arow + k + g * 16 + 8);
            // B fragments: 32x16, lane holds col n, K = g*16 + j (contiguous)
            const uint4* wzp = (const uint4*)(wzrow + k + g * 16);
            wz.q[0] = wzp[0]; wz.q[1] = wzp[1];
            const uint4* whp = (const uint4*)(whrow + k + g * 16);
            wh.q[0] = whp[0]; wh.q[1] = whp[1];

            accZ = __builtin_amdgcn_wmma_f32_16x16x32_bf16(
                       false, a.v, false, wz.v, (short)0, accZ, false, false);
            accU = __builtin_amdgcn_wmma_f32_16x16x32_bf16(
                       false, a.v, false, wh.v, (short)0, accU, false, false);
        }

        // Activation: C layout -> lane holds column n, rows M = r + 8*g
        float gf[8], uf[8];
#pragma unroll
        for (int r = 0; r < 8; ++r) {
            float z = accZ[r] + bzv;
            gf[r] = 1.0f / (1.0f + __expf(-z));
            uf[r] = accU[r] + bhv;
        }

        // Scan: pass 1 (rows 0-7, valid in low lanes), carry handoff via
        // lane-permute, pass 2 (rows 8-15, valid in high lanes).
        float c = carry;
        float hlow[8];
#pragma unroll
        for (int r = 0; r < 8; ++r) { c += gf[r] * (uf[r] - c); hlow[r] = c; }

        float clow = __shfl(c, n, 32);          // row-7 carry from low partner
        float c2 = clow;
        float hhigh[8];
#pragma unroll
        for (int r = 0; r < 8; ++r) { c2 += gf[r] * (uf[r] - c2); hhigh[r] = c2; }

        const bool hi = (g != 0);
        carry = __shfl(hi ? c2 : c, 16 + n, 32); // row-15 carry -> all lanes

        unsigned short* orow =
            hsb + (size_t)b * T_ * H_ + (size_t)(t0 + 8 * g) * H_ + h0 + n;
#pragma unroll
        for (int r = 0; r < 8; ++r)
            orow[(size_t)r * H_] = f32_bf16_rne(hi ? hhigh[r] : hlow[r]);
    }
}

// ---------------------------------------------------------------------------
// Phase 2: out = hs @ Wo^T + bo.  Each wave computes a 16x64 strip: one A
// fragment reused across 4 wmma per K-step. Block = 256 threads = 8 waves.
// ---------------------------------------------------------------------------
__global__ __launch_bounds__(256)
void out_gemm(const unsigned short* __restrict__ hsb,   // [B*T, H] bf16
              const unsigned short* __restrict__ Wob,   // [O, H]  bf16
              const float* __restrict__ bo,
              float* __restrict__ out) {                // [B*T, O] f32
    const int lane = threadIdx.x & 31;
    const int wave = threadIdx.x >> 5;
    const int n    = lane & 15;
    const int g    = lane >> 4;

    const int colGroups = O_ / 64;                       // 16
    const int job      = blockIdx.x * 8 + wave;
    const int rowTile  = job / colGroups;
    const int colGroup = job % colGroups;
    const int row0 = rowTile * 16;
    const int col0 = colGroup * 64;

    v8f acc0 = {}, acc1 = {}, acc2 = {}, acc3 = {};
    const unsigned short* arow = hsb + (size_t)(row0 + n) * H_;
    const unsigned short* w0 = Wob + (size_t)(col0 + n) * H_;
    const unsigned short* w1 = w0 + (size_t)16 * H_;
    const unsigned short* w2 = w1 + (size_t)16 * H_;
    const unsigned short* w3 = w2 + (size_t)16 * H_;

    for (int k = 0; k < H_; k += 32) {
        Frag a, b0, b1, b2, b3;
        a.q[0] = *(const uint4*)(arow + k + g * 16);
        a.q[1] = *(const uint4*)(arow + k + g * 16 + 8);
        const int ko = k + g * 16;
        b0.q[0] = ((const uint4*)(w0 + ko))[0]; b0.q[1] = ((const uint4*)(w0 + ko))[1];
        b1.q[0] = ((const uint4*)(w1 + ko))[0]; b1.q[1] = ((const uint4*)(w1 + ko))[1];
        b2.q[0] = ((const uint4*)(w2 + ko))[0]; b2.q[1] = ((const uint4*)(w2 + ko))[1];
        b3.q[0] = ((const uint4*)(w3 + ko))[0]; b3.q[1] = ((const uint4*)(w3 + ko))[1];

        acc0 = __builtin_amdgcn_wmma_f32_16x16x32_bf16(
                   false, a.v, false, b0.v, (short)0, acc0, false, false);
        acc1 = __builtin_amdgcn_wmma_f32_16x16x32_bf16(
                   false, a.v, false, b1.v, (short)0, acc1, false, false);
        acc2 = __builtin_amdgcn_wmma_f32_16x16x32_bf16(
                   false, a.v, false, b2.v, (short)0, acc2, false, false);
        acc3 = __builtin_amdgcn_wmma_f32_16x16x32_bf16(
                   false, a.v, false, b3.v, (short)0, acc3, false, false);
    }

    float* obase = out + (size_t)(row0 + 8 * g) * O_ + col0 + n;
    const float bo0 = bo[col0 + n];
    const float bo1 = bo[col0 + 16 + n];
    const float bo2 = bo[col0 + 32 + n];
    const float bo3 = bo[col0 + 48 + n];
#pragma unroll
    for (int r = 0; r < 8; ++r) {
        float* op = obase + (size_t)r * O_;
        op[0]  = acc0[r] + bo0;
        op[16] = acc1[r] + bo1;
        op[32] = acc2[r] + bo2;
        op[48] = acc3[r] + bo3;
    }
}

// ---------------------------------------------------------------------------
extern "C" void kernel_launch(void* const* d_in, const int* in_sizes, int n_in,
                              void* d_out, int out_size, void* d_ws, size_t ws_size,
                              hipStream_t stream) {
    const float* xs = (const float*)d_in[0];
    const float* Wz = (const float*)d_in[1];
    const float* bz = (const float*)d_in[2];
    const float* Wh = (const float*)d_in[3];
    const float* bh = (const float*)d_in[4];
    const float* Wo = (const float*)d_in[5];
    const float* bo = (const float*)d_in[6];
    float* out = (float*)d_out;

    // Workspace layout (bf16 staging buffers), ~134 MB total
    char* ws = (char*)d_ws;
    size_t off = 0;
    unsigned short* xs_bf = (unsigned short*)(ws + off); off += (size_t)B_ * T_ * I_ * 2;
    unsigned short* Wz_bf = (unsigned short*)(ws + off); off += (size_t)H_ * I_ * 2;
    unsigned short* Wh_bf = (unsigned short*)(ws + off); off += (size_t)H_ * I_ * 2;
    unsigned short* Wo_bf = (unsigned short*)(ws + off); off += (size_t)O_ * H_ * 2;
    unsigned short* hs_bf = (unsigned short*)(ws + off); off += (size_t)B_ * T_ * H_ * 2;

    // Phase 0: conversions
    {
        int nxs = B_ * T_ * I_;
        cvt_f32_to_bf16<<<(nxs + 255) / 256, 256, 0, stream>>>(xs, xs_bf, nxs);
        int nw = H_ * I_;
        cvt_f32_to_bf16<<<(nw + 255) / 256, 256, 0, stream>>>(Wz, Wz_bf, nw);
        cvt_f32_to_bf16<<<(nw + 255) / 256, 256, 0, stream>>>(Wh, Wh_bf, nw);
        int nwo = O_ * H_;
        cvt_f32_to_bf16<<<(nwo + 255) / 256, 256, 0, stream>>>(Wo, Wo_bf, nwo);
    }

    // Phase 1: fused gate/update GEMM + recurrence scan
    {
        dim3 grid(H_ / 64, B_);
        fused_zu_scan<<<grid, 128, 0, stream>>>(xs_bf, Wz_bf, Wh_bf, bz, bh, hs_bf);
    }

    // Phase 2: output GEMM
    {
        int numJobs = (B_ * T_ / 16) * (O_ / 64);   // 32768
        out_gemm<<<numJobs / 8, 256, 0, stream>>>(hs_bf, Wo_bf, bo, out);
    }
}
